// MoeLayer_52398601011377
// MI455X (gfx1250) — compile-verified, matrix operand order checked by reading, not run on hardware
//
#include <hip/hip_runtime.h>
#include <hip/hip_bf16.h>

// ---------------------------------------------------------------------------
// MoE (top-2 of 8 experts), D=1024, H=4096, T=8192 (f32 in/out).
// Sparse gathered-token design; bf16 WMMA (v_wmma_f32_16x16x32_bf16) core.
// Weights pre-converted to bf16 in workspace when ws_size allows (fast path);
// otherwise converted on the fly (fallback, identical numerics).
// ---------------------------------------------------------------------------

#define N_EMBD   1024
#define FF_HID   4096
#define NUM_EXP  8
#define CHUNK_H  512      // H processed per LDS chunk
#define BM       32       // tokens per block tile

typedef __attribute__((ext_vector_type(16))) __bf16 v16bf;
typedef __attribute__((ext_vector_type(8)))  float  v8f;

union Frag16 {
    v16bf v;
    unsigned short u[16];
    uint4 q[2];
};

__device__ __forceinline__ unsigned short f32_to_bf16(float f) {
    unsigned int x = __float_as_uint(f);
    unsigned int r = x + 0x7FFFu + ((x >> 16) & 1u);   // round-to-nearest-even
    return (unsigned short)(r >> 16);
}

// A/B fragment for 16-bit WMMA from a row-major bf16 tile (LDS or global).
// Lane holds row (lane&15); its 16 elements are K runs [kb+8g .. +7] and
// [kb+16+8g .. +7], g = lane>>4.  Two 16B loads, no conversion.
__device__ __forceinline__ v16bf frag_bf16(const unsigned short* rowptr, int kb, int g) {
    Frag16 f;
    f.q[0] = *(const uint4*)(rowptr + kb + 8 * g);        // *_load_b128
    f.q[1] = *(const uint4*)(rowptr + kb + 16 + 8 * g);   // *_load_b128
    return f.v;
}

// Fallback: same fragment sourced from row-major f32 global memory.
__device__ __forceinline__ v16bf frag_f32(const float* rowptr, int kb, int g) {
    const float* p0 = rowptr + kb + 8 * g;
    float4 a = *(const float4*)(p0);
    float4 b = *(const float4*)(p0 + 4);
    float4 c = *(const float4*)(p0 + 16);
    float4 d = *(const float4*)(p0 + 20);
    float t[16] = {a.x, a.y, a.z, a.w, b.x, b.y, b.z, b.w,
                   c.x, c.y, c.z, c.w, d.x, d.y, d.z, d.w};
    Frag16 f;
#pragma unroll
    for (int i = 0; i < 16; ++i) f.u[i] = f32_to_bf16(t[i]);
    return f.v;
}

// CW = cached bf16 weights available.
template <bool CW>
__device__ __forceinline__ v16bf load_wfrag(const float* fp, const unsigned short* bp,
                                            size_t rowoff, int kb, int g) {
    if constexpr (CW) return frag_bf16(bp + rowoff, kb, g);
    else              return frag_f32(fp + rowoff, kb, g);
}

// ---------------------------------------------------------------------------
// f32 -> bf16 bulk conversion (8 elements / thread: 2x b128 load, 1x b128 store)
// ---------------------------------------------------------------------------
__global__ void cvt_bf16_kernel(const float* __restrict__ src,
                                unsigned short* __restrict__ dst, size_t n8) {
    size_t i = (size_t)blockIdx.x * blockDim.x + threadIdx.x;
    if (i >= n8) return;
    size_t off = i * 8;
    float4 a = *(const float4*)(src + off);
    float4 b = *(const float4*)(src + off + 4);
    float t[8] = {a.x, a.y, a.z, a.w, b.x, b.y, b.z, b.w};
    union { uint4 q; unsigned short u[8]; } o;
#pragma unroll
    for (int j = 0; j < 8; ++j) o.u[j] = f32_to_bf16(t[j]);
    *(uint4*)(dst + off) = o.q;
}

// ---------------------------------------------------------------------------
// Kernel 1: gating. One wave per token: 8 logits, shfl reduce, top-2 softmax,
// append token to per-expert gather lists. Row order within a list cannot
// change numerics (M rows independent; final combine is a 2-term commutative
// float sum into a zeroed output).
// ---------------------------------------------------------------------------
__global__ void gate_kernel(const float* __restrict__ x,
                            const float* __restrict__ wg,
                            int* __restrict__ counts,
                            int* __restrict__ idxbuf,
                            float* __restrict__ wgtbuf,
                            int T) {
    const int wavesPerBlock = blockDim.x >> 5;
    const int tok  = blockIdx.x * wavesPerBlock + (threadIdx.x >> 5);
    const int lane = threadIdx.x & 31;
    if (tok >= T) return;

    const float* xr = x + (size_t)tok * N_EMBD;
    float xv[N_EMBD / 32];
#pragma unroll
    for (int i = 0; i < N_EMBD / 32; ++i) xv[i] = xr[lane + 32 * i];

    float logits[NUM_EXP];
#pragma unroll
    for (int e = 0; e < NUM_EXP; ++e) {
        const float* wr = wg + e * N_EMBD;
        float s = 0.f;
#pragma unroll
        for (int i = 0; i < N_EMBD / 32; ++i) s += xv[i] * wr[lane + 32 * i];
#pragma unroll
        for (int off = 16; off > 0; off >>= 1) s += __shfl_xor(s, off, 32);
        logits[e] = s;
    }

    if (lane == 0) {
        int b0 = 0;
#pragma unroll
        for (int e = 1; e < NUM_EXP; ++e) if (logits[e] > logits[b0]) b0 = e;
        int b1 = (b0 == 0) ? 1 : 0;
#pragma unroll
        for (int e = 0; e < NUM_EXP; ++e) {
            if (e == b0) continue;
            if (logits[e] > logits[b1]) b1 = e;
        }
        float m  = logits[b0];
        float e0 = __expf(logits[b0] - m);
        float e1 = __expf(logits[b1] - m);
        float inv = __builtin_amdgcn_rcpf(e0 + e1);
        int p0 = atomicAdd(&counts[b0], 1);
        idxbuf[b0 * T + p0] = tok;  wgtbuf[b0 * T + p0] = e0 * inv;
        int p1 = atomicAdd(&counts[b1], 1);
        idxbuf[b1 * T + p1] = tok;  wgtbuf[b1 * T + p1] = e1 * inv;
    }
}

// ---------------------------------------------------------------------------
// Kernel 2: fused expert FFN on gathered tokens.
//   out[tok] += wgt * ( silu(x w1^T) * (x w3^T) ) w2^T
// 512 threads = 16 waves (wave32). Per 512-wide H chunk:
//   phase 1: h chunk -> LDS (paired WMMAs share the A fragment)
//   phase 2: y += h * w2^T into register accumulators (8 x v8f per wave).
// ---------------------------------------------------------------------------
template <bool CW>
__global__ __launch_bounds__(512)
void moe_ffn_kernel(const float* __restrict__ x,
                    const float* __restrict__ w1f,
                    const float* __restrict__ w2f,
                    const float* __restrict__ w3f,
                    const unsigned short* __restrict__ w1b,
                    const unsigned short* __restrict__ w2b,
                    const unsigned short* __restrict__ w3b,
                    const int* __restrict__ counts,
                    const int* __restrict__ idxbuf,
                    const float* __restrict__ wgtbuf,
                    float* __restrict__ out,
                    int T) {
    __shared__ alignas(16) unsigned short xs_s[BM * N_EMBD];  // 64 KB bf16
    __shared__ alignas(16) unsigned short h_s[BM * CHUNK_H];  // 32 KB bf16
    __shared__ int   tok_s[BM];
    __shared__ float wt_s[BM];

    const int e     = blockIdx.y;
    const int cnt   = counts[e];
    const int tile0 = blockIdx.x * BM;
    if (tile0 >= cnt) return;                      // block-sparse skip
    const int rows = (cnt - tile0 < BM) ? (cnt - tile0) : BM;

    const int tid  = threadIdx.x;
    const int lane = tid & 31;
    const int w    = tid >> 5;        // wave 0..15
    const int n15  = lane & 15;
    const int g    = lane >> 4;

    if (tid < BM) {
        int p = tile0 + tid;
        if (p < cnt) { tok_s[tid] = idxbuf[(size_t)e * T + p]; wt_s[tid] = wgtbuf[(size_t)e * T + p]; }
        else         { tok_s[tid] = 0;                         wt_s[tid] = 0.f; }
    }
    __syncthreads();

    // Stage gathered x tile as bf16 in LDS (zero-pad invalid rows).
    for (int i = tid; i < BM * N_EMBD; i += 512) {
        int r = i >> 10, c = i & (N_EMBD - 1);
        float v = (r < rows) ? x[(size_t)tok_s[r] * N_EMBD + c] : 0.f;
        xs_s[i] = f32_to_bf16(v);
    }
    __syncthreads();

    const size_t eWD = (size_t)e * FF_HID * N_EMBD;
    const float*          w1fe = w1f + eWD;
    const float*          w3fe = w3f + eWD;
    const float*          w2fe = w2f + eWD;
    const unsigned short* w1be = CW ? (w1b + eWD) : nullptr;
    const unsigned short* w3be = CW ? (w3b + eWD) : nullptr;
    const unsigned short* w2be = CW ? (w2b + eWD) : nullptr;

    const int mt    = w & 1;            // token half-tile owned by this wave
    const int nbase = (w >> 1) * 128;   // 128-wide output slab (8 N tiles)

    const v8f vzero = {0.f, 0.f, 0.f, 0.f, 0.f, 0.f, 0.f, 0.f};
    v8f y[8];
#pragma unroll
    for (int t = 0; t < 8; ++t) y[t] = vzero;

    const unsigned short* xrow = xs_s + (mt * 16 + n15) * N_EMBD;
    const unsigned short* hrow = h_s  + (mt * 16 + n15) * CHUNK_H;

    for (int c0 = 0; c0 < FF_HID; c0 += CHUNK_H) {
        // ---- phase 1: h[BM, CHUNK_H] = silu(x w1^T) * (x w3^T) ----
#pragma unroll
        for (int it = 0; it < 4; ++it) {
            const int ht   = (w >> 1) + 8 * it;            // hidden tile in chunk
            const int habs = c0 + ht * 16 + n15;           // this lane's B row
            const size_t ro = (size_t)habs * N_EMBD;
            v8f a1 = vzero, a3 = vzero;
            for (int kb = 0; kb < N_EMBD; kb += 32) {
                if constexpr (CW) {
                    __builtin_prefetch(w1be + ro + kb + 128, 0, 1);  // global_prefetch_b8
                    __builtin_prefetch(w3be + ro + kb + 128, 0, 1);
                } else {
                    __builtin_prefetch(w1fe + ro + kb + 64, 0, 1);
                    __builtin_prefetch(w3fe + ro + kb + 64, 0, 1);
                }
                v16bf af = frag_bf16(xrow, kb, g);
                v16bf b1 = load_wfrag<CW>(w1fe, w1be, ro, kb, g);
                v16bf b3 = load_wfrag<CW>(w3fe, w3be, ro, kb, g);
                a1 = __builtin_amdgcn_wmma_f32_16x16x32_bf16(
                        false, af, false, b1, (short)0, a1, false, false);
                a3 = __builtin_amdgcn_wmma_f32_16x16x32_bf16(
                        false, af, false, b3, (short)0, a3, false, false);
            }
#pragma unroll
            for (int r = 0; r < 8; ++r) {                  // C/D: M = r + 8g, N = n15
                float s1 = a1[r], s3 = a3[r];
                float hv = s1 * __builtin_amdgcn_rcpf(1.f + __expf(-s1)) * s3;  // silu*gate
                int tr = mt * 16 + 8 * g + r;
                int hc = ht * 16 + n15;
                h_s[tr * CHUNK_H + hc] = f32_to_bf16(hv);
            }
        }
        __syncthreads();

        // ---- phase 2: y[BM, N_EMBD] += h * w2^T (register accumulators) ----
        for (int ks = 0; ks < CHUNK_H; ks += 32) {
            v16bf af = frag_bf16(hrow, ks, g);
#pragma unroll
            for (int nt = 0; nt < 8; ++nt) {
                const size_t ro = (size_t)(nbase + nt * 16 + n15) * FF_HID + c0;
                v16bf bfr = load_wfrag<CW>(w2fe, w2be, ro, ks, g);
                y[nt] = __builtin_amdgcn_wmma_f32_16x16x32_bf16(
                           false, af, false, bfr, (short)0, y[nt], false, false);
            }
        }
        __syncthreads();   // h_s reused next chunk
    }

    // Scale by routing weight and accumulate (2-term commutative float sum).
#pragma unroll
    for (int nt = 0; nt < 8; ++nt) {
#pragma unroll
        for (int r = 0; r < 8; ++r) {
            int tr = mt * 16 + 8 * g + r;
            if (tr < rows) {
                int col = nbase + nt * 16 + n15;
                atomicAdd(&out[(size_t)tok_s[tr] * N_EMBD + col], wt_s[tr] * y[nt][r]);
            }
        }
    }
}

// ---------------------------------------------------------------------------
extern "C" void kernel_launch(void* const* d_in, const int* in_sizes, int n_in,
                              void* d_out, int out_size, void* d_ws, size_t ws_size,
                              hipStream_t stream) {
    const float* x  = (const float*)d_in[0];
    const float* wg = (const float*)d_in[1];
    const float* w1 = (const float*)d_in[2];
    const float* w2 = (const float*)d_in[3];
    const float* w3 = (const float*)d_in[4];
    float* out = (float*)d_out;

    const int T = in_sizes[0] / N_EMBD;   // 8192 tokens

    // workspace layout: [counts 16i] [idx E*T i32] [wgt E*T f32] [bf16 W cache]
    int*   counts = (int*)d_ws;
    int*   idxbuf = counts + 16;
    float* wgtbuf = (float*)(idxbuf + (size_t)NUM_EXP * T);

    const size_t wElems = (size_t)NUM_EXP * FF_HID * N_EMBD;   // per weight tensor
    uintptr_t cacheBase = ((uintptr_t)(wgtbuf + (size_t)NUM_EXP * T) + 255) & ~(uintptr_t)255;
    unsigned short* w1b = (unsigned short*)cacheBase;
    unsigned short* w3b = w1b + wElems;
    unsigned short* w2b = w3b + wElems;
    const size_t needBytes = (uintptr_t)((unsigned char*)(w2b + wElems) - (unsigned char*)d_ws);
    const bool cached = (ws_size >= needBytes);

    hipMemsetAsync(d_out, 0, (size_t)out_size * sizeof(float), stream);
    hipMemsetAsync(counts, 0, 16 * sizeof(int), stream);

    if (cached) {
        const size_t n8 = wElems / 8;
        const int cthr = 256;
        const int cblk = (int)((n8 + cthr - 1) / cthr);
        cvt_bf16_kernel<<<cblk, cthr, 0, stream>>>(w1, w1b, n8);
        cvt_bf16_kernel<<<cblk, cthr, 0, stream>>>(w3, w3b, n8);
        cvt_bf16_kernel<<<cblk, cthr, 0, stream>>>(w2, w2b, n8);
    }

    gate_kernel<<<(T + 7) / 8, 256, 0, stream>>>(x, wg, counts, idxbuf, wgtbuf, T);

    dim3 grid((T + BM - 1) / BM, NUM_EXP);
    if (cached) {
        moe_ffn_kernel<true><<<grid, 512, 0, stream>>>(
            x, w1, w2, w3, w1b, w2b, w3b, counts, idxbuf, wgtbuf, out, T);
    } else {
        moe_ffn_kernel<false><<<grid, 512, 0, stream>>>(
            x, w1, w2, w3, nullptr, nullptr, nullptr, counts, idxbuf, wgtbuf, out, T);
    }
}
